// CCAMDec_21595095564741
// MI455X (gfx1250) — matrix operand order repaired and checked by reading, not compiled, not run on hardware
//
#include <hip/hip_runtime.h>
#include <hip/hip_bf16.h>

typedef float  v8f   __attribute__((ext_vector_type(8)));
typedef __bf16 v16bf __attribute__((ext_vector_type(16)));
typedef unsigned int v4u __attribute__((ext_vector_type(4)));
typedef int v8i __attribute__((ext_vector_type(8)));
typedef int v4i __attribute__((ext_vector_type(4)));

#define B_DIM 8
#define C_DIM 512
#define K_DIM 512
#define N_DIM 4096
#define XS 36   // padded LDS row stride in floats (32 data + 4 pad), 16B aligned

union Frag {
  v16bf v;
  unsigned short s[16];
  uint4 q[2];
};

__device__ __forceinline__ unsigned short bf16_bits(float f) {
  return __builtin_bit_cast(unsigned short, (__bf16)f);
}

// ---------------------------------------------------------------------------
// TDM: DMA a 2D tile (32 floats wide x tile_rows tall, row stride N_DIM) from
// global into LDS, inserting 4 pad DWORDs after every 32 DWORDs so the LDS
// tile has row stride XS=36 floats.  D# packed per CDNA5 ISA §8.3/8.4.
// This toolchain exposes the 6-arg builtin: (g0, g1, g2, g3, g_extra, cpol).
// ---------------------------------------------------------------------------
__device__ __forceinline__ void tdm_load_2d(unsigned lds_addr,
                                            unsigned long long ga,
                                            int tile_rows) {
  v4u g0 = { 1u,                                    // count=1, user descriptor
             lds_addr,                              // lds_addr [63:32]
             (unsigned)ga,                          // global_addr lo [95:64]
             ((unsigned)(ga >> 32) & 0x01FFFFFFu)   // global_addr hi [120:96]
                 | (2u << 30) };                    // type=2 ("image")
  v8i g1 = {
      (int)((2u << 16)      // data_size = 4 bytes
            | (1u << 20)    // pad_enable
            | (4u << 22)    // pad_interval: code 4 -> every 32 DWORDs
            | (3u << 25)),  // pad_amount:   code 3 -> 4 DWORDs
      (int)(((unsigned)N_DIM & 0xFFFFu) << 16),          // tensor_dim0 lo16
      (int)(((unsigned)N_DIM >> 16) | 0xFFFF0000u),      // dim0 hi / dim1 lo
      (int)(32u << 16),                                  // dim1 hi=0, tile_dim0=32
      tile_rows,                                         // tile_dim1 (tile_dim2=0)
      (int)N_DIM,                                        // tensor_dim0_stride lo32
      0, 0 };                                            // stride hi / dim1_stride
  v4i gz4 = { 0, 0, 0, 0 };
  v8i gz8 = { 0, 0, 0, 0, 0, 0, 0, 0 };
  __builtin_amdgcn_tensor_load_to_lds(g0, g1, gz4, gz4, gz8, 0);
}

// ---------------------------------------------------------------------------
// Kernel 1: energy[b,c,k] = sum_n X[b,c,n] * Y[b,k,n]   (bf16 WMMA, f32 acc)
// Grid: B * (C/128) * (K/64); block 256 = 8 waves; wave tile 32x32 (2x2 WMMA).
// X (128x32) and Y (64x32) chunk tiles are DMAed into LDS by the Tensor Data
// Mover (wave 0 issues; TENSORcnt), double-buffered to overlap copy/compute.
// ---------------------------------------------------------------------------
__global__ __launch_bounds__(256) void energy_kernel(
    const float* __restrict__ x, const float* __restrict__ y,
    float* __restrict__ energy) {
  __shared__ float xt[2][128 * XS];
  __shared__ float yt[2][64 * XS];

  const int bid  = blockIdx.x;
  const int tk   = bid & 7;          // K/64 tiles
  const int tm   = (bid >> 3) & 3;   // C/128 tiles
  const int b    = bid >> 5;
  const int wave = threadIdx.x >> 5;
  const int lane = threadIdx.x & 31;
  const int wm = wave & 3;           // 4 waves along M
  const int wn = wave >> 2;          // 2 waves along output-K
  const int m_base = tm * 128 + wm * 32;
  const int c_base = tk * 64  + wn * 32;
  const int lr = lane & 15;
  const int lh = lane >> 4;

  const float* xbase = x + ((size_t)(b * C_DIM) + tm * 128) * N_DIM;
  const float* ybase = y + ((size_t)(b * K_DIM) + tk * 64) * N_DIM;

  v8f acc[2][2] = {};

  if (wave == 0) {   // prologue: DMA chunk 0 into buffer 0
    tdm_load_2d((unsigned)(size_t)&xt[0][0],
                (unsigned long long)(const void*)(xbase), 128);
    tdm_load_2d((unsigned)(size_t)&yt[0][0],
                (unsigned long long)(const void*)(ybase), 64);
  }

  for (int kc = 0; kc < N_DIM; kc += 32) {
    const int ib = (kc >> 5) & 1;
    if (wave == 0) {
      if (kc + 32 < N_DIM) {   // DMA next chunk into the other buffer
        tdm_load_2d((unsigned)(size_t)&xt[ib ^ 1][0],
                    (unsigned long long)(const void*)(xbase + kc + 32), 128);
        tdm_load_2d((unsigned)(size_t)&yt[ib ^ 1][0],
                    (unsigned long long)(const void*)(ybase + kc + 32), 64);
        __builtin_amdgcn_s_wait_tensorcnt(2);   // current chunk's DMAs done
      } else {
        __builtin_amdgcn_s_wait_tensorcnt(0);
      }
    }
    __syncthreads();

    Frag fa[2], fb[2];
    // A fragments: lane lr = row, lh selects K-half.
#pragma unroll
    for (int i = 0; i < 2; ++i) {
      const float* p = &xt[ib][(wm * 32 + i * 16 + lr) * XS + lh * 16];
      float4 fv[4];
      fv[0] = *(const float4*)(p);
      fv[1] = *(const float4*)(p + 4);
      fv[2] = *(const float4*)(p + 8);
      fv[3] = *(const float4*)(p + 12);
      const float* g = (const float*)fv;
#pragma unroll
      for (int t = 0; t < 16; ++t) fa[i].v[t] = (__bf16)g[t];
    }
    // B fragments: lane lr = output column (row of Y tile).
#pragma unroll
    for (int j = 0; j < 2; ++j) {
      const float* p = &yt[ib][(wn * 32 + j * 16 + lr) * XS + lh * 16];
      float4 fv[4];
#pragma unroll
      for (int t = 0; t < 4; ++t) fv[t] = *(const float4*)(p + 4 * t);
      const float* g = (const float*)fv;
#pragma unroll
      for (int t = 0; t < 16; ++t) fb[j].v[t] = (__bf16)g[t];
    }
#pragma unroll
    for (int i = 0; i < 2; ++i)
#pragma unroll
      for (int j = 0; j < 2; ++j)
        acc[i][j] = __builtin_amdgcn_wmma_f32_16x16x32_bf16(
            false, fa[i].v, false, fb[j].v, (short)0, acc[i][j], false, false);

    __syncthreads();   // tile consumed before next DMA overwrites it
  }

  float* e = energy + (size_t)(b * C_DIM) * K_DIM;
#pragma unroll
  for (int i = 0; i < 2; ++i)
#pragma unroll
    for (int v = 0; v < 8; ++v) {
      const int row = m_base + i * 16 + lh * 8 + v;
#pragma unroll
      for (int j = 0; j < 2; ++j) {
        const int col = c_base + j * 16 + lr;
        e[(size_t)row * K_DIM + col] = acc[i][j][v];
      }
    }
}

// ---------------------------------------------------------------------------
// Kernel 2: attn = softmax(max - e) over K == softmax(-e).
// Stable: p = exp(min - e) / sum. One wave per row; output packed bf16 pairs.
// ---------------------------------------------------------------------------
__global__ __launch_bounds__(256) void softmax_kernel(
    const float* __restrict__ energy, unsigned int* __restrict__ attn) {
  const int row  = blockIdx.x * 8 + (threadIdx.x >> 5);
  const int lane = threadIdx.x & 31;
  const float* e = energy + (size_t)row * K_DIM + lane * 16;

  float4 fv[4];
#pragma unroll
  for (int t = 0; t < 4; ++t) fv[t] = *(const float4*)(e + 4 * t);
  const float* vals = (const float*)fv;

  float mn = vals[0];
#pragma unroll
  for (int t = 1; t < 16; ++t) mn = fminf(mn, vals[t]);
#pragma unroll
  for (int off = 16; off >= 1; off >>= 1)
    mn = fminf(mn, __shfl_xor(mn, off, 32));

  float p[16];
  float sum = 0.f;
#pragma unroll
  for (int t = 0; t < 16; ++t) { p[t] = __expf(mn - vals[t]); sum += p[t]; }
#pragma unroll
  for (int off = 16; off >= 1; off >>= 1)
    sum += __shfl_xor(sum, off, 32);
  const float inv = 1.0f / sum;

  unsigned int packed[8];
#pragma unroll
  for (int t = 0; t < 8; ++t)
    packed[t] = (unsigned int)bf16_bits(p[2 * t] * inv) |
                ((unsigned int)bf16_bits(p[2 * t + 1] * inv) << 16);

  unsigned int* dst = attn + (size_t)row * (K_DIM / 2) + lane * 8;
  *(uint4*)(dst)     = *(const uint4*)&packed[0];
  *(uint4*)(dst + 4) = *(const uint4*)&packed[4];
}

// ---------------------------------------------------------------------------
// Kernel 3: out[b,c,n] = x[b,c,n] + scale * sum_k attn[b,c,k] * Y[b,k,n]
// Grid: B * (C/128) * (N/64). Y k-chunks (32x64 f32, 8 KB) are staged into a
// double-buffered LDS tile with gfx1250 async-to-LDS copies (ASYNCcnt), so
// the copy of chunk i+1 overlaps the WMMAs on chunk i.
// ---------------------------------------------------------------------------
__global__ __launch_bounds__(256) void out_kernel(
    const float* __restrict__ x, const float* __restrict__ y,
    const unsigned int* __restrict__ attn, const float* __restrict__ scale_p,
    float* __restrict__ out) {
  __shared__ float ytile[2][32 * 64];   // [buf][kk][n - n_tile0]

  const int bid  = blockIdx.x;
  const int tn   = bid & 63;          // N/64 tiles
  const int tm   = (bid >> 6) & 3;    // C/128 tiles
  const int b    = bid >> 8;
  const int tid  = threadIdx.x;
  const int wave = tid >> 5;
  const int lane = tid & 31;
  const int wm = wave & 3, wn = wave >> 2;
  const int m_base = tm * 128 + wm * 32;
  const int n_tile = tn * 64;
  const int n_base = n_tile + wn * 32;
  const int lr = lane & 15, lh = lane >> 4;

  const unsigned int* ar =
      attn + ((size_t)(b * C_DIM) + m_base + lr) * (K_DIM / 2);
  const float* yb = y + (size_t)(b * K_DIM) * N_DIM;

  // Cooperative async staging: thread -> 32 contiguous bytes of the tile.
  const int ld_row = tid >> 3;        // 0..31 (kk)
  const int ld_col = (tid & 7) * 8;   // 0..56 (floats)

  auto issue_copy = [&](int kc, int buf) {
    const float* src = yb + (size_t)(kc + ld_row) * N_DIM + n_tile + ld_col;
    unsigned long long ga = (unsigned long long)(const void*)src;
    unsigned lo = (unsigned)(size_t)&ytile[buf][ld_row * 64 + ld_col];
    asm volatile("global_load_async_to_lds_b128 %0, %1, off"
                 :: "v"(lo), "v"(ga) : "memory");
    asm volatile("global_load_async_to_lds_b128 %0, %1, off offset:16"
                 :: "v"(lo), "v"(ga) : "memory");
  };

  v8f acc[2][2] = {};

  issue_copy(0, 0);   // prologue

  for (int kc = 0; kc < K_DIM; kc += 32) {
    const int ib = (kc >> 5) & 1;
    if (kc + 32 < K_DIM) {
      issue_copy(kc + 32, ib ^ 1);
      asm volatile("s_wait_asynccnt 0x2" ::: "memory");  // current chunk done
    } else {
      asm volatile("s_wait_asynccnt 0x0" ::: "memory");
    }
    __syncthreads();

    Frag fa[2], fb[2];
#pragma unroll
    for (int i = 0; i < 2; ++i) {
      const unsigned int* p = ar + (size_t)i * 16 * (K_DIM / 2) + (kc >> 1);
      fa[i].q[0] = *(const uint4*)(p + lh * 4);       // K = lh*8 + 0..7
      fa[i].q[1] = *(const uint4*)(p + 8 + lh * 4);   // K = 16 + lh*8 + 0..7
    }
#pragma unroll
    for (int j = 0; j < 2; ++j) {
      const int col = wn * 32 + j * 16 + lr;
#pragma unroll
      for (int t = 0; t < 16; ++t)
        fb[j].v[t] = (__bf16)ytile[ib][(lh * 8 + (t & 7) + (t >> 3) * 16) * 64 + col];
    }
#pragma unroll
    for (int i = 0; i < 2; ++i)
#pragma unroll
      for (int j = 0; j < 2; ++j)
        acc[i][j] = __builtin_amdgcn_wmma_f32_16x16x32_bf16(
            false, fa[i].v, false, fb[j].v, (short)0, acc[i][j], false, false);

    __syncthreads();   // tile consumed before next chunk overwrites it
  }

  const float scale = scale_p[0];
#pragma unroll
  for (int i = 0; i < 2; ++i)
#pragma unroll
    for (int v = 0; v < 8; ++v) {
      const int row = m_base + i * 16 + lh * 8 + v;
#pragma unroll
      for (int j = 0; j < 2; ++j) {
        const int n = n_base + j * 16 + lr;
        const size_t idx = ((size_t)(b * C_DIM) + row) * N_DIM + n;
        out[idx] = x[idx] + scale * acc[i][j][v];
      }
    }
}

extern "C" void kernel_launch(void* const* d_in, const int* in_sizes, int n_in,
                              void* d_out, int out_size, void* d_ws, size_t ws_size,
                              hipStream_t stream) {
  const float* x     = (const float*)d_in[0];
  const float* y     = (const float*)d_in[1];
  const float* scale = (const float*)d_in[2];
  float* out = (float*)d_out;

  // Workspace: energy f32 [B,C,K] = 8 MB, then packed-bf16 attn = 4 MB.
  float* energy = (float*)d_ws;
  unsigned int* attn =
      (unsigned int*)((char*)d_ws + (size_t)B_DIM * C_DIM * K_DIM * sizeof(float));

  energy_kernel<<<B_DIM * (C_DIM / 128) * (K_DIM / 64), 256, 0, stream>>>(
      x, y, energy);
  softmax_kernel<<<(B_DIM * C_DIM) / 8, 256, 0, stream>>>(energy, attn);
  out_kernel<<<B_DIM * (C_DIM / 128) * (N_DIM / 64), 256, 0, stream>>>(
      x, y, attn, scale, out);
}
